// DBTransformer_28492813041701
// MI455X (gfx1250) — matrix operand order repaired and checked by reading, not compiled, not run on hardware
//
#include <hip/hip_runtime.h>
#include <math.h>

// Problem constants (from reference)
#define NA_ 40000
#define NB_ 40000
#define EDGES_ 150000
#define C_ 64
#define T_ 4
#define TC_ 256   // T*C
#define OUT_ 32

typedef __bf16 bf16;
typedef bf16  v16bf __attribute__((ext_vector_type(16)));
typedef float v8f   __attribute__((ext_vector_type(8)));

// ---------------------------------------------------------------------------
// WMMA fragment loaders (CDNA5 wave32, V_WMMA_F32_16X16X32_BF16)
// A (16x32 bf16): lane L holds row m=L%16, group g=L/16,
//   element j -> K = 16g + j (consecutive, same order as B)
// B (32x16 bf16): lane L holds col n=L%16, group g=L/16,
//   element j -> K = 16g + j (consecutive)
// D (16x16 f32): lane L col n=L%16, reg r -> row m = r + 8g
// ---------------------------------------------------------------------------
__device__ __forceinline__ v16bf load_A_frag(const float* __restrict__ row32, int g) {
  const float* p0 = row32 + g * 16;       // K = 16g .. 16g+7
  const float* p1 = row32 + g * 16 + 8;   // K = 16g+8 .. 16g+15
  v16bf a;
#pragma unroll
  for (int j = 0; j < 8; ++j) a[j] = (bf16)p0[j];
#pragma unroll
  for (int j = 0; j < 8; ++j) a[8 + j] = (bf16)p1[j];
  return a;
}

__device__ __forceinline__ v16bf load_B_frag(const float* __restrict__ wrun16) {
  v16bf b;
#pragma unroll
  for (int j = 0; j < 16; ++j) b[j] = (bf16)wrun16[j];
  return b;
}

// Preload per-lane bias registers (bias[n] for n = nt*16 + (lane&15))
template <int NT>
__device__ __forceinline__ void load_bias(const float* __restrict__ b, int nl,
                                          float (&r)[NT]) {
#pragma unroll
  for (int nt = 0; nt < NT; ++nt) r[nt] = b[nt * 16 + nl];
}

// Out[m][n] = sum_k A[m][k] * W[n][k] + bias[n]
// A: LDS 16x64 f32 rows; W: global [NT*16][64] f32 (loop-invariant -> B frags
// stay resident in VGPRs); bias: preloaded registers.
// ROWMODE==1: b_proj remap -> store only rows m<8, 0..3 -> 4+m, 4..7 -> 8+m
template <int NT, bool RELU, int ROWMODE>
__device__ __forceinline__ void matmul16(const float* __restrict__ A,
                                         const float* __restrict__ W,
                                         const float (&bias)[NT],
                                         float* __restrict__ Out, int ostride,
                                         int lane) {
  const int nl = lane & 15;
  const int g  = lane >> 4;
#pragma unroll
  for (int nt = 0; nt < NT; ++nt) {
    const int n = nt * 16 + nl;
    v8f acc = {0.f, 0.f, 0.f, 0.f, 0.f, 0.f, 0.f, 0.f};
#pragma unroll
    for (int kt = 0; kt < 2; ++kt) {
      v16bf a = load_A_frag(A + nl * 64 + kt * 32, g);
      v16bf b = load_B_frag(W + (size_t)n * 64 + kt * 32 + g * 16);
      acc = __builtin_amdgcn_wmma_f32_16x16x32_bf16(
          false, a, false, b, (short)0, acc, false, false);
    }
    const float bv = bias[nt];
#pragma unroll
    for (int r = 0; r < 8; ++r) {
      int m = r + 8 * g;
      float v = acc[r] + bv;
      if (RELU) v = fmaxf(v, 0.0f);
      if (ROWMODE == 1) {
        if (g) continue;                 // rows 8..15 unused for b_proj
        m = (r < 4) ? (r + 4) : (r + 8); // edge0 xj -> rows 4..7, edge1 -> 12..15
      }
      Out[m * ostride + n] = v;
    }
  }
}

// X[m] = LayerNorm(X[m] + R[m]) * g + b   (rows 0..15; lanes 0..15, LDS params)
__device__ __forceinline__ void layernorm_inplace(float* __restrict__ X,
                                                  const float* __restrict__ R,
                                                  const float* __restrict__ gam,
                                                  const float* __restrict__ bet,
                                                  int lane) {
  if (lane < 16) {
    const int m = lane;
    float s1 = 0.f, s2 = 0.f;
#pragma unroll 4
    for (int c = 0; c < 64; ++c) {
      float v = X[m * 64 + c] + R[m * 64 + c];
      s1 += v; s2 += v * v;
    }
    const float mean = s1 * (1.0f / 64.0f);
    const float var  = s2 * (1.0f / 64.0f) - mean * mean;
    const float rs   = rsqrtf(var + 1e-5f);
#pragma unroll 4
    for (int c = 0; c < 64; ++c) {
      float v = X[m * 64 + c] + R[m * 64 + c];
      X[m * 64 + c] = (v - mean) * rs * gam[c] + bet[c];
    }
  }
}

// Prefetch one 1KB node row into cache (global_prefetch_b8); 8 lanes x 128B
__device__ __forceinline__ void prefetch_row(const float* __restrict__ p, int lane) {
  if (lane < 8) __builtin_prefetch(p + lane * 32, 0, 3);
}

// ---------------------------------------------------------------------------
// Edge-conv kernel: 2 waves/block, each wave handles 2 edges per iteration
// (16 sequence rows = full WMMA M). Scatter-sum via global fp32 atomics.
// Edge indices pipelined one iteration ahead; next rows prefetched into L2/WGP$.
// ---------------------------------------------------------------------------
__global__ void __launch_bounds__(64)
conv_kernel(const float* __restrict__ x_src, const float* __restrict__ x_dst,
            const int* __restrict__ ei, int nE,
            const float* __restrict__ Wb,   const float* __restrict__ bb,
            const float* __restrict__ Wqkv, const float* __restrict__ bqkv,
            const float* __restrict__ Wo,   const float* __restrict__ bo,
            const float* __restrict__ g1,   const float* __restrict__ be1,
            const float* __restrict__ W1,   const float* __restrict__ bf1,
            const float* __restrict__ W2,   const float* __restrict__ bf2,
            const float* __restrict__ g2,   const float* __restrict__ be2,
            float* __restrict__ acc, float* __restrict__ cnt) {
  __shared__ float sbuf[2][6144]; // per wave: X[1024] QKV[3072] TMP[1024] T2[1024]
  __shared__ float sln[256];      // g1 | be1 | g2 | be2

  const int tid  = threadIdx.x;
  const int wave = tid >> 5;
  const int lane = tid & 31;
  const int nl   = lane & 15;

  // Stage LayerNorm params once per block
  sln[tid]       = g1[tid];
  sln[64 + tid]  = be1[tid];
  sln[128 + tid] = g2[tid];
  sln[192 + tid] = be2[tid];

  // Preload biases into registers (loop-invariant)
  float bb_r[4], bq_r[12], bo_r[4], b1_r[4], b2_r[4];
  load_bias<4>(bb, nl, bb_r);
  load_bias<12>(bqkv, nl, bq_r);
  load_bias<4>(bo, nl, bo_r);
  load_bias<4>(bf1, nl, b1_r);
  load_bias<4>(bf2, nl, b2_r);

  float* X   = &sbuf[wave][0];
  float* QKV = X + 1024;
  float* TMP = X + 4096;
  float* T2  = X + 5120;

  const int wgid   = blockIdx.x * 2 + wave;
  const int nwaves = gridDim.x * 2;
  const int npairs = (nE + 1) >> 1;
  const int iters  = (npairs + nwaves - 1) / nwaves;

  // Indices for iteration 0
  bool v0c = (wgid < npairs) && (2 * wgid < nE);
  bool v1c = (wgid < npairs) && (2 * wgid + 1 < nE);
  int s0c = v0c ? ei[2 * wgid] : 0, d0c = v0c ? ei[nE + 2 * wgid] : 0;
  int s1c = v1c ? ei[2 * wgid + 1] : 0, d1c = v1c ? ei[nE + 2 * wgid + 1] : 0;

  __syncthreads();

  for (int it = 0; it < iters; ++it) {
    // ---- pipeline: fetch next iteration's indices, prefetch next rows ----
    const int pn = wgid + (it + 1) * nwaves;
    const bool v0n = (it + 1 < iters) && (pn < npairs) && (2 * pn < nE);
    const bool v1n = (it + 1 < iters) && (pn < npairs) && (2 * pn + 1 < nE);
    const int s0n = v0n ? ei[2 * pn] : 0, d0n = v0n ? ei[nE + 2 * pn] : 0;
    const int s1n = v1n ? ei[2 * pn + 1] : 0, d1n = v1n ? ei[nE + 2 * pn + 1] : 0;
    if (v0n) {
      prefetch_row(x_src + (size_t)s0n * TC_, lane);
      prefetch_row(x_dst + (size_t)d0n * TC_, lane);
    }
    if (v1n) {
      prefetch_row(x_src + (size_t)s1n * TC_, lane);
      prefetch_row(x_dst + (size_t)d1n * TC_, lane);
    }

    // ---- gather: xi -> X rows {0..3, 8..11}; raw xj -> TMP rows {0..3,4..7}
#pragma unroll
    for (int ee = 0; ee < 2; ++ee) {
      const int si = ee ? s1c : s0c, di = ee ? d1c : d0c;
#pragma unroll
      for (int j = 0; j < 8; ++j) {
        const int f = lane + j * 32, t = f >> 6, c = f & 63;
        X[(ee * 8 + t) * 64 + c]   = x_dst[(size_t)di * TC_ + f];
        TMP[(ee * 4 + t) * 64 + c] = x_src[(size_t)si * TC_ + f];
      }
    }
    __syncthreads();

    // b_proj(xj) -> X rows {4..7, 12..15}
    matmul16<4, false, 1>(TMP, Wb, bb_r, X, 64, lane);
    __syncthreads();

    // QKV: [16,64] @ [192,64]^T -> [16,192]
    matmul16<12, false, 0>(X, Wqkv, bq_r, QKV, 192, lane);
    __syncthreads();

    // Attention (VALU): 2 edges x 4 heads x 8 rows = 64 combos, 2/lane -> TMP
#pragma unroll
    for (int cc = 0; cc < 2; ++cc) {
      const int co = lane + cc * 32;
      const int e  = co >> 5;
      const int h  = (co & 31) >> 3;
      const int s  = co & 7;
      const float* qrow = QKV + (e * 8 + s) * 192 + h * 16;
      float sc[8];
      float mx = -1e30f;
#pragma unroll
      for (int t = 0; t < 8; ++t) {
        const float* krow = QKV + (e * 8 + t) * 192 + 64 + h * 16;
        float d = 0.f;
#pragma unroll
        for (int u = 0; u < 16; ++u) d += qrow[u] * krow[u];
        sc[t] = d * 0.25f; // 1/sqrt(DH)
        mx = fmaxf(mx, sc[t]);
      }
      float sum = 0.f;
#pragma unroll
      for (int t = 0; t < 8; ++t) { sc[t] = __expf(sc[t] - mx); sum += sc[t]; }
      const float inv = 1.0f / sum;
#pragma unroll
      for (int u = 0; u < 16; ++u) {
        float o = 0.f;
#pragma unroll
        for (int t = 0; t < 8; ++t)
          o += sc[t] * QKV[(e * 8 + t) * 192 + 128 + h * 16 + u];
        TMP[(e * 8 + s) * 64 + h * 16 + u] = o * inv;
      }
    }
    __syncthreads();

    // out-proj + residual LN1
    matmul16<4, false, 0>(TMP, Wo, bo_r, T2, 64, lane);
    __syncthreads();
    layernorm_inplace(X, T2, sln, sln + 64, lane);
    __syncthreads();

    // FF: relu(X@W1^T+bf1) @ W2^T + bf2, residual LN2
    matmul16<4, true, 0>(X, W1, b1_r, TMP, 64, lane);
    __syncthreads();
    matmul16<4, false, 0>(TMP, W2, b2_r, T2, 64, lane);
    __syncthreads();
    layernorm_inplace(X, T2, sln + 128, sln + 192, lane);
    __syncthreads();

    // Scatter-sum first T=4 tokens of each edge
#pragma unroll
    for (int ee = 0; ee < 2; ++ee) {
      const bool vv = ee ? v1c : v0c;
      const int di  = ee ? d1c : d0c;
      if (vv) {
#pragma unroll
        for (int j = 0; j < 8; ++j) {
          const int f = lane + j * 32, t = f >> 6, c = f & 63;
          atomicAdd(&acc[(size_t)di * TC_ + f], X[(ee * 8 + t) * 64 + c]);
        }
        if (lane == 0) atomicAdd(&cnt[di], 1.0f);
      }
    }
    __syncthreads();

    // rotate pipelined indices
    s0c = s0n; d0c = d0n; s1c = s1n; d1c = d1n; v0c = v0n; v1c = v1n;
  }
}

// ---------------------------------------------------------------------------
// Final kernel: one wave per node; lane = output class.
// h[o] = sum_{t,c} out[n,t,c]*Wout[o,c] + T*bout[o]; softmax over o.
// ---------------------------------------------------------------------------
__global__ void __launch_bounds__(128)
out_kernel(const float* __restrict__ acc0, const float* __restrict__ cnt0,
           const float* __restrict__ acc1, const float* __restrict__ cnt1,
           const float* __restrict__ Wout, const float* __restrict__ bout,
           float* __restrict__ out) {
  __shared__ float sWT[64 * 32]; // transposed Wout: [c][o] (bank-conflict-free)
  __shared__ float sb[32];
  __shared__ float sv[4][256];

  const int tid = threadIdx.x;
  for (int i = tid; i < 2048; i += 128) {
    const int c = i >> 5, o = i & 31;
    sWT[i] = Wout[o * 64 + c];
  }
  if (tid < 32) sb[tid] = bout[tid];
  __syncthreads();

  const int wave = tid >> 5, lane = tid & 31;
  const int node = blockIdx.x * 4 + wave; // NA divisible by 4

  const float c0 = cnt0[node], c1 = cnt1[node];
  const float i0 = (c0 > 0.f) ? (1.0f / c0) : 0.0f;
  const float i1 = (c1 > 0.f) ? (1.0f / c1) : 0.0f;

  float* v = sv[wave];
#pragma unroll
  for (int j = 0; j < 8; ++j) {
    const int f = lane + j * 32;
    v[f] = 0.5f * (acc0[(size_t)node * TC_ + f] * i0 +
                   acc1[(size_t)node * TC_ + f] * i1);
  }
  __syncthreads();

  float h = (float)T_ * sb[lane]; // bout summed over T tokens
  for (int f = 0; f < 256; ++f)
    h += v[f] * sWT[(f & 63) * 32 + lane];

  float m = h;
#pragma unroll
  for (int off = 16; off > 0; off >>= 1) m = fmaxf(m, __shfl_xor(m, off, 32));
  const float ex = __expf(h - m);
  float s = ex;
#pragma unroll
  for (int off = 16; off > 0; off >>= 1) s += __shfl_xor(s, off, 32);

  out[(size_t)node * OUT_ + lane] = ex / s;
}

__global__ void zero_kernel(float* __restrict__ p, size_t n) {
  size_t i = (size_t)blockIdx.x * blockDim.x + threadIdx.x;
  const size_t stride = (size_t)gridDim.x * blockDim.x;
  for (; i < n; i += stride) p[i] = 0.0f;
}

// ---------------------------------------------------------------------------
extern "C" void kernel_launch(void* const* d_in, const int* in_sizes, int n_in,
                              void* d_out, int out_size, void* d_ws, size_t ws_size,
                              hipStream_t stream) {
  (void)in_sizes; (void)n_in; (void)out_size; (void)ws_size;

  const float* x_A  = (const float*)d_in[0];
  const float* x_B  = (const float*)d_in[1];
  const int*   e_ba = (const int*)d_in[2];
  const int*   e_aa = (const int*)d_in[3];
  const float* Wb   = (const float*)d_in[4];
  const float* bb   = (const float*)d_in[5];
  const float* Wqkv = (const float*)d_in[6];
  const float* bqkv = (const float*)d_in[7];
  const float* Wo   = (const float*)d_in[8];
  const float* bo   = (const float*)d_in[9];
  const float* g1   = (const float*)d_in[10];
  const float* be1  = (const float*)d_in[11];
  const float* W1   = (const float*)d_in[12];
  const float* bf1  = (const float*)d_in[13];
  const float* W2   = (const float*)d_in[14];
  const float* bf2  = (const float*)d_in[15];
  const float* g2   = (const float*)d_in[16];
  const float* be2  = (const float*)d_in[17];
  const float* Wout = (const float*)d_in[18];
  const float* bout = (const float*)d_in[19];

  float* ws   = (float*)d_ws;
  float* acc0 = ws;                                  // NA*256
  float* acc1 = ws + (size_t)NA_ * TC_;              // NA*256
  float* cnt0 = ws + (size_t)2 * NA_ * TC_;          // NA
  float* cnt1 = cnt0 + NA_;                          // NA

  const size_t nzero = (size_t)2 * NA_ * TC_ + 2 * NA_;
  zero_kernel<<<1024, 256, 0, stream>>>(ws, nzero);

  // Edge type 0: ('B','ba','A')
  conv_kernel<<<2048, 64, 0, stream>>>(
      x_B, x_A, e_ba, EDGES_,
      Wb, bb, Wqkv, bqkv, Wo, bo, g1, be1,
      W1, bf1, W2, bf2, g2, be2, acc0, cnt0);

  // Edge type 1: ('A','aa','A') — weight slice k=1
  conv_kernel<<<2048, 64, 0, stream>>>(
      x_A, x_A, e_aa, EDGES_,
      Wb + 64 * 64, bb + 64, Wqkv + 192 * 64, bqkv + 192,
      Wo + 64 * 64, bo + 64, g1 + 64, be1 + 64,
      W1 + 64 * 64, bf1 + 64, W2 + 64 * 64, bf2 + 64,
      g2 + 64, be2 + 64, acc1, cnt1);

  out_kernel<<<NA_ / 4, 128, 0, stream>>>(acc0, cnt0, acc1, cnt1,
                                          Wout, bout, (float*)d_out);
}
